// self_attn_mini_3350074491550
// MI455X (gfx1250) — matrix operand order repaired and checked by reading, not compiled, not run on hardware
//
#include <hip/hip_runtime.h>

typedef __attribute__((ext_vector_type(16))) __bf16 v16bf;
typedef __attribute__((ext_vector_type(8)))  __bf16 bf16x8;
typedef __attribute__((ext_vector_type(8)))  float  v8f;
typedef __attribute__((ext_vector_type(4)))  int    i32x4;

#define D_ 1024
#define S_ 2048
#define B_ 8
#define LDT 40              // padded LDS row stride (80B = 20 dwords; conflict-free)
#define TILE_ELEMS (128 * LDT)

#if __has_builtin(__builtin_amdgcn_global_load_async_to_lds_b128)
#define HAS_ASYNC_LDS 1
// ROCm-header-style address-space casts: builtin takes (global int4*, lds int4*, imm off, imm cpol)
#define GPTR(p) ((__attribute__((address_space(1))) i32x4*)(p))
#define LPTR(p) ((__attribute__((address_space(3))) i32x4*)(p))
#define ACP16(gp, lp, OFF) \
  __builtin_amdgcn_global_load_async_to_lds_b128(GPTR(gp), LPTR(lp), OFF, 0)
#else
#define HAS_ASYNC_LDS 0
#endif

__device__ inline void wait_async8() {
#if __has_builtin(__builtin_amdgcn_s_wait_asynccnt)
  __builtin_amdgcn_s_wait_asynccnt(8);
#else
  asm volatile("s_wait_asynccnt 8" ::: "memory");
#endif
}
__device__ inline void wait_async0() {
#if __has_builtin(__builtin_amdgcn_s_wait_asynccnt)
  __builtin_amdgcn_s_wait_asynccnt(0);
#else
  asm volatile("s_wait_asynccnt 0" ::: "memory");
#endif
}

__device__ inline v8f wmma_bf16(v16bf a, v16bf b, v8f c) {
  return __builtin_amdgcn_wmma_f32_16x16x32_bf16(
      /*neg_a=*/false, a, /*neg_b=*/false, b,
      /*c_mod=*/(short)0, c, /*reuse_a=*/false, /*reuse_b=*/false);
}

__device__ inline v8f zero_v8f() {
  v8f z;
#pragma unroll
  for (int e = 0; e < 8; ++e) z[e] = 0.0f;
  return z;
}

// ---- Fragment loaders (ISA 7.12.2 layouts) --------------------------------
__device__ inline v16bf a_frag_bf16(const __bf16* base, int ld, int lane) {
  const __bf16* r = base + (size_t)(lane & 15) * ld + ((lane >> 4) << 3);
  union { v16bf v; bf16x8 h[2]; } u;
  u.h[0] = *(const bf16x8*)r;
  u.h[1] = *(const bf16x8*)(r + 16);
  return u.v;
}
__device__ inline v16bf b_frag_bf16(const __bf16* base, int ld, int lane) {
  const __bf16* r = base + (size_t)(lane & 15) * ld + ((lane >> 4) << 3);
  union { v16bf v; bf16x8 h[2]; } u;
  u.h[0] = *(const bf16x8*)r;
  u.h[1] = *(const bf16x8*)(r + 16);
  return u.v;
}

// 16 WMMAs on one staged 128x32 A-tile / 128x32 B-tile pair
__device__ inline void tile_mma(const __bf16* As, const __bf16* Bs, int m0w,
                                int n0w, int lane, v8f c[4][4]) {
  v16bf a[4], bfr[4];
#pragma unroll
  for (int i = 0; i < 4; ++i) a[i]   = a_frag_bf16(As + (m0w + 16 * i) * LDT, LDT, lane);
#pragma unroll
  for (int j = 0; j < 4; ++j) bfr[j] = b_frag_bf16(Bs + (n0w + 16 * j) * LDT, LDT, lane);
#pragma unroll
  for (int i = 0; i < 4; ++i)
#pragma unroll
    for (int j = 0; j < 4; ++j) c[i][j] = wmma_bf16(a[i], bfr[j], c[i][j]);
}

// ---- Cooperative 128x128 block GEMM core, K-steps of 32 through LDS -------
// Ablk: [blockRow0, 0] of A (bf16, ld=lda); Bblk: [blockCol0, 0] of SRC where
// B = SRC^T. 128 threads = 4 waves; each wave owns a 64x64 quadrant.
// As/Bs are double buffers of 2*TILE_ELEMS each.
__device__ inline void gemm_core_128(const __bf16* __restrict__ Ablk, int lda,
                                     const __bf16* __restrict__ Bblk, int ldb,
                                     int K, __bf16* As, __bf16* Bs,
                                     v8f c[4][4]) {
  const int t = threadIdx.x;
  const int lane = t & 31;
  const int w = t >> 5;
  const int m0w = (w & 1) * 64;
  const int n0w = (w >> 1) * 64;
  const int nk = K >> 5;
  const __bf16* ga = Ablk + (size_t)t * lda;  // this thread's A row
  const __bf16* gb = Bblk + (size_t)t * ldb;  // this thread's B row

#if HAS_ASYNC_LDS
  __bf16* la = As + t * LDT;
  __bf16* lb = Bs + t * LDT;
  // prologue: stage tile 0 into buffer 0 (8 async b128 per thread)
  ACP16(ga, la, 0);  ACP16(ga, la, 16); ACP16(ga, la, 32); ACP16(ga, la, 48);
  ACP16(gb, lb, 0);  ACP16(gb, lb, 16); ACP16(gb, lb, 32); ACP16(gb, lb, 48);
  for (int kt = 0; kt < nk; ++kt) {
    const int cur = kt & 1;
    if (kt + 1 < nk) {  // stage next tile into the other buffer, then wait cur
      const __bf16* gan = ga + (kt + 1) * 32;
      const __bf16* gbn = gb + (kt + 1) * 32;
      __bf16* lan = la + (cur ^ 1) * TILE_ELEMS;
      __bf16* lbn = lb + (cur ^ 1) * TILE_ELEMS;
      ACP16(gan, lan, 0); ACP16(gan, lan, 16); ACP16(gan, lan, 32); ACP16(gan, lan, 48);
      ACP16(gbn, lbn, 0); ACP16(gbn, lbn, 16); ACP16(gbn, lbn, 32); ACP16(gbn, lbn, 48);
      wait_async8();  // in-order: >=8 left means current tile's 8 are done
    } else {
      wait_async0();
    }
    __syncthreads();  // current tile visible to all waves
    tile_mma(As + cur * TILE_ELEMS, Bs + cur * TILE_ELEMS, m0w, n0w, lane, c);
    __syncthreads();  // tile consumed before it is re-staged next iteration
  }
#else
  for (int kt = 0; kt < nk; ++kt) {
    const int k0 = kt << 5;
    const bf16x8* pga = (const bf16x8*)(ga + k0);
    const bf16x8* pgb = (const bf16x8*)(gb + k0);
    bf16x8 ra0 = pga[0], ra1 = pga[1], ra2 = pga[2], ra3 = pga[3];
    bf16x8 rb0 = pgb[0], rb1 = pgb[1], rb2 = pgb[2], rb3 = pgb[3];
    if (k0 + 32 < K) {
      __builtin_prefetch((const void*)(ga + k0 + 32), 0, 3);
      __builtin_prefetch((const void*)(gb + k0 + 32), 0, 3);
    }
    __syncthreads();
    bf16x8* sa = (bf16x8*)(As + t * LDT);
    sa[0] = ra0; sa[1] = ra1; sa[2] = ra2; sa[3] = ra3;
    bf16x8* sb = (bf16x8*)(Bs + t * LDT);
    sb[0] = rb0; sb[1] = rb1; sb[2] = rb2; sb[3] = rb3;
    __syncthreads();
    tile_mma(As, Bs, m0w, n0w, lane, c);
  }
#endif
}

// ---- fp32 -> bf16 bulk convert (8 elems/thread) ---------------------------
__global__ __launch_bounds__(256) void cvt_kernel(const float* __restrict__ src,
                                                  __bf16* __restrict__ dst, int n8) {
  int i = blockIdx.x * 256 + threadIdx.x;
  if (i >= n8) return;
  const float* s = src + (size_t)i * 8;
  bf16x8 t;
#pragma unroll
  for (int r = 0; r < 8; ++r) t[r] = (__bf16)s[r];
  *(bf16x8*)(dst + (size_t)i * 8) = t;
}

// ---- Projections: OUT = Xb @ Wb^T (+pos for K), bf16 out ------------------
// mode 0: Q row-major; 1: K' = +pos row-major; 2: V transposed as vt[b][e][s]
__global__ __launch_bounds__(128) void proj_kernel(
    const __bf16* __restrict__ xb, const __bf16* __restrict__ Wb,
    const float* __restrict__ pos, __bf16* __restrict__ outp, int mode) {
  __shared__ __bf16 As[2 * TILE_ELEMS];
  __shared__ __bf16 Bs[2 * TILE_ELEMS];
  const int lane = threadIdx.x & 31;
  const int w = threadIdx.x >> 5;
  const int m0 = blockIdx.y * 128 + (w & 1) * 64;
  const int n0 = blockIdx.x * 128 + (w >> 1) * 64;

  v8f c[4][4];
#pragma unroll
  for (int i = 0; i < 4; ++i)
#pragma unroll
    for (int j = 0; j < 4; ++j) c[i][j] = zero_v8f();

  gemm_core_128(xb + (size_t)blockIdx.y * 128 * D_, D_,
                Wb + (size_t)blockIdx.x * 128 * D_, D_, D_, As, Bs, c);

  const int col = lane & 15;
  const int rsel = (lane >> 4) << 3;
#pragma unroll
  for (int i = 0; i < 4; ++i)
#pragma unroll
    for (int j = 0; j < 4; ++j) {
      const int e = n0 + 16 * j + col;
      const int mb = m0 + 16 * i + rsel;
      if (mode == 2) {
        const int bb = mb >> 11, s = mb & (S_ - 1);  // 8 consecutive s, one batch
        bf16x8 tv;
#pragma unroll
        for (int r = 0; r < 8; ++r) tv[r] = (__bf16)c[i][j][r];
        *(bf16x8*)(outp + ((size_t)bb * D_ + e) * S_ + s) = tv;
      } else {
#pragma unroll
        for (int r = 0; r < 8; ++r) {
          const size_t m = (size_t)(mb + r);
          float v = c[i][j][r];
          if (mode == 1) v += pos[m * D_ + e];
          outp[m * D_ + e] = (__bf16)v;
        }
      }
    }
}

// ---- Scores: S = Q @ K'^T / sqrt(D), fp32 ---------------------------------
__global__ __launch_bounds__(128) void scores_kernel(
    const __bf16* __restrict__ q, const __bf16* __restrict__ kp,
    float* __restrict__ sc) {
  __shared__ __bf16 As[2 * TILE_ELEMS];
  __shared__ __bf16 Bs[2 * TILE_ELEMS];
  const int lane = threadIdx.x & 31;
  const int w = threadIdx.x >> 5;
  const int b = blockIdx.z;
  const __bf16* qb = q + (size_t)b * S_ * D_;
  const __bf16* kb = kp + (size_t)b * S_ * D_;
  float* scb = sc + (size_t)b * S_ * S_;
  const int m0 = blockIdx.y * 128 + (w & 1) * 64;
  const int n0 = blockIdx.x * 128 + (w >> 1) * 64;

  v8f c[4][4];
#pragma unroll
  for (int i = 0; i < 4; ++i)
#pragma unroll
    for (int j = 0; j < 4; ++j) c[i][j] = zero_v8f();

  gemm_core_128(qb + (size_t)blockIdx.y * 128 * D_, D_,
                kb + (size_t)blockIdx.x * 128 * D_, D_, D_, As, Bs, c);

  const float scale = 0.03125f;  // 1/sqrt(1024)
  const int col = lane & 15;
  const int rsel = (lane >> 4) << 3;
#pragma unroll
  for (int i = 0; i < 4; ++i)
#pragma unroll
    for (int j = 0; j < 4; ++j) {
      const int n = n0 + 16 * j + col;
      const int mb = m0 + 16 * i + rsel;
#pragma unroll
      for (int r = 0; r < 8; ++r)
        scb[(size_t)(mb + r) * S_ + n] = c[i][j][r] * scale;
    }
}

// ---- Row softmax; writes bf16 P in place over its own fp32 row ------------
__global__ __launch_bounds__(256) void softmax_kernel(float* __restrict__ sc) {
  __shared__ float buf[S_];
  __shared__ float red[256];
  const int row = blockIdx.x;
  float* srow = sc + (size_t)row * S_;
  const int t = threadIdx.x;

  float mx = -3.0e38f;
  for (int i = t; i < S_; i += 256) {
    float v = srow[i];
    buf[i] = v;
    mx = fmaxf(mx, v);
  }
  red[t] = mx;
  __syncthreads();
  for (int s = 128; s > 0; s >>= 1) {
    if (t < s) red[t] = fmaxf(red[t], red[t + s]);
    __syncthreads();
  }
  mx = red[0];
  __syncthreads();

  float sum = 0.0f;
  for (int i = t; i < S_; i += 256) {
    float e = __expf(buf[i] - mx);
    buf[i] = e;
    sum += e;
  }
  red[t] = sum;
  __syncthreads();
  for (int s = 128; s > 0; s >>= 1) {
    if (t < s) red[t] += red[t + s];
    __syncthreads();
  }
  const float inv = 1.0f / red[0];
  __syncthreads();

  __bf16* prow = (__bf16*)srow;  // first half of own 8KB row; no cross-block alias
  for (int i = t; i < S_; i += 256) prow[i] = (__bf16)(buf[i] * inv);
}

// ---- O = P @ V; P bf16 at row stride 4096 elems; V pre-transposed ---------
__global__ __launch_bounds__(128) void av_kernel(
    const char* __restrict__ scores_bytes, const __bf16* __restrict__ vt,
    float* __restrict__ out) {
  __shared__ __bf16 As[2 * TILE_ELEMS];
  __shared__ __bf16 Bs[2 * TILE_ELEMS];
  const int lane = threadIdx.x & 31;
  const int w = threadIdx.x >> 5;
  const int b = blockIdx.z;
  const __bf16* pb = (const __bf16*)(scores_bytes + (size_t)b * S_ * S_ * 4);
  const int ldP = S_ * 2;  // bf16 elems between P rows (8 KB)
  const __bf16* vtb = vt + (size_t)b * D_ * S_;
  const int m0 = blockIdx.y * 128 + (w & 1) * 64;
  const int n0 = blockIdx.x * 128 + (w >> 1) * 64;  // over D

  v8f c[4][4];
#pragma unroll
  for (int i = 0; i < 4; ++i)
#pragma unroll
    for (int j = 0; j < 4; ++j) c[i][j] = zero_v8f();

  gemm_core_128(pb + (size_t)blockIdx.y * 128 * ldP, ldP,
                vtb + (size_t)blockIdx.x * 128 * S_, S_, S_, As, Bs, c);

  const int col = lane & 15;
  const int rsel = (lane >> 4) << 3;
#pragma unroll
  for (int i = 0; i < 4; ++i)
#pragma unroll
    for (int j = 0; j < 4; ++j) {
      const int e = n0 + 16 * j + col;
      const int mb = m0 + 16 * i + rsel;
#pragma unroll
      for (int r = 0; r < 8; ++r)
        out[((size_t)b * S_ + (mb + r)) * D_ + e] = c[i][j][r];
    }
}

extern "C" void kernel_launch(void* const* d_in, const int* in_sizes, int n_in,
                              void* d_out, int out_size, void* d_ws, size_t ws_size,
                              hipStream_t stream) {
  const float* x   = (const float*)d_in[0];
  const float* pos = (const float*)d_in[1];
  const float* Wq  = (const float*)d_in[2];
  const float* Wk  = (const float*)d_in[3];
  const float* Wv  = (const float*)d_in[4];
  float* out = (float*)d_out;

  // ws layout (MB): [q 0..32][kp 32..64][vt 64..96][xb 96..128][Wb 128..134]
  // scores (128MB fp32) reuses 96..224 once xb/Wb are dead.
  char* ws = (char*)d_ws;
  const size_t MB = 1024 * 1024;
  __bf16* q      = (__bf16*)(ws);
  __bf16* kp     = (__bf16*)(ws + 32 * MB);
  __bf16* vt     = (__bf16*)(ws + 64 * MB);
  __bf16* xb     = (__bf16*)(ws + 96 * MB);
  __bf16* wqb    = (__bf16*)(ws + 128 * MB);
  __bf16* wkb    = (__bf16*)(ws + 130 * MB);
  __bf16* wvb    = (__bf16*)(ws + 132 * MB);
  char*   scores = ws + 96 * MB;

  const int xn8 = (B_ * S_ * D_) / 8;
  const int wn8 = (D_ * D_) / 8;
  cvt_kernel<<<dim3(xn8 / 256), dim3(256), 0, stream>>>(x, xb, xn8);
  cvt_kernel<<<dim3(wn8 / 256), dim3(256), 0, stream>>>(Wq, wqb, wn8);
  cvt_kernel<<<dim3(wn8 / 256), dim3(256), 0, stream>>>(Wk, wkb, wn8);
  cvt_kernel<<<dim3(wn8 / 256), dim3(256), 0, stream>>>(Wv, wvb, wn8);

  dim3 blk(128);
  dim3 pg(D_ / 128, (B_ * S_) / 128, 1);
  proj_kernel<<<pg, blk, 0, stream>>>(xb, wqb, pos, q, 0);
  proj_kernel<<<pg, blk, 0, stream>>>(xb, wkb, pos, kp, 1);
  proj_kernel<<<pg, blk, 0, stream>>>(xb, wvb, pos, vt, 2);

  dim3 sg(S_ / 128, S_ / 128, B_);
  scores_kernel<<<sg, blk, 0, stream>>>(q, kp, (float*)scores);

  softmax_kernel<<<B_ * S_, 256, 0, stream>>>((float*)scores);

  dim3 ag(D_ / 128, S_ / 128, B_);
  av_kernel<<<ag, blk, 0, stream>>>(scores, vt, out);
}
